// CTM_13039520710760
// MI455X (gfx1250) — compile-verified, matrix operand order checked by reading, not run on hardware
//
#include <hip/hip_runtime.h>

// ---------------------------------------------------------------------------
// CTM forward for MI455X (gfx1250, wave32).
// Heavy GEMMs: v_wmma_f32_16x16x32_f16 (f16 operands, f32 accumulate),
// B operands pre-transposed to [N,K] f16 so both A and B tiles stage with
// contiguous vector loads; B staging uses global_load_async_to_lds_b128
// (ASYNCcnt) on full tiles. Elementwise / layernorm / softmax / NLM fused
// into fp32 VALU kernels.
// ---------------------------------------------------------------------------

typedef __attribute__((ext_vector_type(16))) _Float16 v16h;
typedef __attribute__((ext_vector_type(4)))  _Float16 v4h;
typedef __attribute__((ext_vector_type(8)))  float    v8f;
typedef __attribute__((ext_vector_type(4)))  float    v4f;

#define BT_    512      // B*T
#define D_     768
#define MTR_   25       // trace depth
#define V_     32000
#define S_     384      // Sa == So
#define TT_    256      // T (per-batch sequence length)
#define ITERS_ 8

#define BM 128
#define BN 64
#define BK 32
#define PADH 8          // LDS row padding in halfs (keeps rows 16B aligned)

// ---------------------------------------------------------------------------
// WMMA GEMM: C[M,N] = A[M,K](f32) * B[N,K](f16, transposed) (+bias),
// optional f32 and/or f16 outputs, batched via gridDim.z.
// 256 threads = 8 waves (4x2); block tile 128x64; wave tile 32x32 (4 WMMAs).
// ---------------------------------------------------------------------------
__global__ __launch_bounds__(256) void wmma_gemm(
    const float* __restrict__ A, const _Float16* __restrict__ B,
    const float* __restrict__ bias,
    float* __restrict__ C, _Float16* __restrict__ Ch,
    int M, int N, int K, int lda, int ldb, int ldc,
    long sA, long sB, long sC)
{
  __shared__ _Float16 lsa[BM][BK + PADH];   // [m][k]
  __shared__ _Float16 lsb[BN][BK + PADH];   // [n][k]
  const int bz = blockIdx.z;
  A += sA * bz;
  B += sB * bz;

  const int tid  = threadIdx.x;
  const int m0   = blockIdx.y * BM;
  const int n0   = blockIdx.x * BN;
  const int w    = tid >> 5;
  const int lane = tid & 31;
  const int wmb  = (w & 3) * 32;          // wave m base in block tile
  const int wnb  = (w >> 2) * 32;         // wave n base in block tile
  const int hi16 = lane >> 4;
  const int frow = lane & 15;
  const int koff = hi16 * 8;

  const bool fastA = (m0 + BM <= M);
  const bool fastB = (n0 + BN <= N);

  v8f acc00 = {0.f,0.f,0.f,0.f,0.f,0.f,0.f,0.f};
  v8f acc01 = {0.f,0.f,0.f,0.f,0.f,0.f,0.f,0.f};
  v8f acc10 = {0.f,0.f,0.f,0.f,0.f,0.f,0.f,0.f};
  v8f acc11 = {0.f,0.f,0.f,0.f,0.f,0.f,0.f,0.f};

  for (int k0 = 0; k0 < K; k0 += BK) {
    const bool fk = (k0 + BK <= K);

    // ---- stage A tile (f32 -> f16), 128x32 ----
    if (fastA && fk) {
#pragma unroll
      for (int i = 0; i < 4; ++i) {
        int e  = tid + i * 256;
        int r  = e >> 3;                   // 8 float4 per row of 32
        int c4 = (e & 7) * 4;
        v4f va = *(const v4f*)(A + (long)(m0 + r) * lda + k0 + c4);
        v4h hv = { (_Float16)va[0], (_Float16)va[1],
                   (_Float16)va[2], (_Float16)va[3] };
        *(v4h*)&lsa[r][c4] = hv;
      }
    } else {
#pragma unroll
      for (int i = 0; i < 16; ++i) {
        int e = tid + i * 256;
        int r = e >> 5, c = e & 31;
        int gm = m0 + r, gk = k0 + c;
        float va = (gm < M && gk < K) ? A[(long)gm * lda + gk] : 0.f;
        lsa[r][c] = (_Float16)va;
      }
    }

    // ---- stage B tile (f16 [N,K] rows), 64x32 ----
    if (fastB && fk) {
      // one 16-byte async global->LDS copy per lane: 256 * 16B = 64x32 halfs
      int r  = tid >> 2;                   // 4 x 8-half chunks per row
      int c8 = (tid & 3) * 8;
      const _Float16* src = B + (long)(n0 + r) * ldb + k0 + c8;
      unsigned ldsoff = (unsigned)(unsigned long long)&lsb[r][c8];
      asm volatile("global_load_async_to_lds_b128 %0, %1, off"
                   :: "v"(ldsoff), "v"(src) : "memory");
    } else {
#pragma unroll
      for (int i = 0; i < 8; ++i) {
        int e = tid + i * 256;
        int r = e >> 5, c = e & 31;
        int gn = n0 + r, gk = k0 + c;
        lsb[r][c] = (gn < N && gk < K) ? B[(long)gn * ldb + gk]
                                       : (_Float16)0.f;
      }
    }

    if (k0 + BK < K)
      __builtin_prefetch(&A[(long)(m0 + (tid >> 1)) * lda + k0 + BK], 0, 0);

    asm volatile("s_wait_asynccnt 0x0" ::: "memory");
    __syncthreads();

    // Fragment layout (quarter-chunk rule, ISA 7.12.2):
    // lanes 0-15 hold K = 0..7 (halfs 0-7) and 16..23 (halfs 8-15);
    // lanes 16-31 hold K = 8..15 and 24..31. Same mapping for A and B.
    v16h a0, a1, b0, b1;
#pragma unroll
    for (int j = 0; j < 16; ++j) {
      int kk = koff + (j & 7) + ((j & 8) ? 16 : 0);
      a0[j] = lsa[wmb + frow][kk];
      a1[j] = lsa[wmb + 16 + frow][kk];
      b0[j] = lsb[wnb + frow][kk];
      b1[j] = lsb[wnb + 16 + frow][kk];
    }
    acc00 = __builtin_amdgcn_wmma_f32_16x16x32_f16(false, a0, false, b0,
                                                   (short)0, acc00, false, false);
    acc01 = __builtin_amdgcn_wmma_f32_16x16x32_f16(false, a0, false, b1,
                                                   (short)0, acc01, false, false);
    acc10 = __builtin_amdgcn_wmma_f32_16x16x32_f16(false, a1, false, b0,
                                                   (short)0, acc10, false, false);
    acc11 = __builtin_amdgcn_wmma_f32_16x16x32_f16(false, a1, false, b1,
                                                   (short)0, acc11, false, false);
    __syncthreads();
  }

  // ---- epilogue: 4 16x16 tiles per wave ----
  auto store_tile = [&](const v8f& acc, int mt, int nt) {
    int gn = nt + frow;
    float bv = (bias && gn < N) ? bias[gn] : 0.f;
#pragma unroll
    for (int r = 0; r < 8; ++r) {
      int gm = mt + hi16 * 8 + r;        // C layout: VGPR r -> M=r (+8 hi lanes)
      if (gm < M && gn < N) {
        float vv = acc[r] + bv;
        long base = sC * bz + (long)gm * ldc + gn;
        if (C)  C[base]  = vv;
        if (Ch) Ch[base] = (_Float16)vv;
      }
    }
  };
  store_tile(acc00, m0 + wmb,      n0 + wnb);
  store_tile(acc01, m0 + wmb,      n0 + wnb + 16);
  store_tile(acc10, m0 + wmb + 16, n0 + wnb);
  store_tile(acc11, m0 + wmb + 16, n0 + wnb + 16);
}

// ---------------------------------------------------------------------------
// Transposing f32 -> f16 weight conversion: out[n*K + k] = (f16) in[k*N + n]
// ---------------------------------------------------------------------------
__global__ __launch_bounds__(256) void cvt_f16_T(const float* __restrict__ in,
                                                 _Float16* __restrict__ out,
                                                 int K, int N) {
  __shared__ float tile[32][33];
  int kb = blockIdx.y * 32, nb = blockIdx.x * 32;
  int tx = threadIdx.x & 31, ty = threadIdx.x >> 5;   // 32 x 8
#pragma unroll
  for (int i = 0; i < 32; i += 8) {
    int k = kb + ty + i, n = nb + tx;
    tile[ty + i][tx] = (k < K && n < N) ? in[(long)k * N + n] : 0.f;
  }
  __syncthreads();
#pragma unroll
  for (int i = 0; i < 32; i += 8) {
    int n = nb + ty + i, k = kb + tx;
    if (n < N && k < K) out[(long)n * K + k] = (_Float16)tile[tx][ty + i];
  }
}

// per-batch transpose of v: in [B*T, D] f16 -> out [B, D, T] f16
__global__ __launch_bounds__(256) void transpose_v_f16(
    const _Float16* __restrict__ in, _Float16* __restrict__ out) {
  __shared__ _Float16 tile[32][33];
  int b = blockIdx.z;
  int tb = blockIdx.y * 32, db = blockIdx.x * 32;
  int tx = threadIdx.x & 31, ty = threadIdx.x >> 5;
#pragma unroll
  for (int i = 0; i < 32; i += 8)
    tile[ty + i][tx] = in[(long)(b * TT_ + tb + ty + i) * D_ + db + tx];
  __syncthreads();
#pragma unroll
  for (int i = 0; i < 32; i += 8)
    out[(long)b * D_ * TT_ + (long)(db + ty + i) * TT_ + tb + tx] =
        tile[tx][ty + i];
}

// ---------------------------------------------------------------------------
// Block reductions (blockDim.x == 256, no divergent returns in callers)
// ---------------------------------------------------------------------------
__device__ __forceinline__ float block_red_sum(float v, float* red) {
  int t = threadIdx.x;
  red[t] = v; __syncthreads();
  for (int s = 128; s > 0; s >>= 1) {
    if (t < s) red[t] += red[t + s];
    __syncthreads();
  }
  float r = red[0]; __syncthreads();
  return r;
}
__device__ __forceinline__ float block_red_max(float v, float* red) {
  int t = threadIdx.x;
  red[t] = v; __syncthreads();
  for (int s = 128; s > 0; s >>= 1) {
    if (t < s) red[t] = fmaxf(red[t], red[t + s]);
    __syncthreads();
  }
  float r = red[0]; __syncthreads();
  return r;
}

// ---------------------------------------------------------------------------
// Small fused kernels
// ---------------------------------------------------------------------------
__global__ void init_state_k(const float* __restrict__ sa, const float* __restrict__ st,
                             float* __restrict__ act, float* __restrict__ trace) {
  long idx = (long)blockIdx.x * blockDim.x + threadIdx.x;
  if (idx >= (long)BT_ * D_) return;
  int d = (int)(idx % D_);
  act[idx] = sa[d];
#pragma unroll
  for (int m = 0; m < MTR_; ++m) trace[idx * MTR_ + m] = st[(long)d * MTR_ + m];
}

__global__ void init_sync_k(const float* __restrict__ sa, const int* __restrict__ il,
                            const int* __restrict__ ir, float* __restrict__ a,
                            float* __restrict__ bsum) {
  long idx = (long)blockIdx.x * blockDim.x + threadIdx.x;
  if (idx >= (long)BT_ * S_) return;
  int s = (int)(idx % S_);
  a[idx] = sa[il[s]] * sa[ir[s]];
  bsum[idx] = 1.f;
}

__global__ void sync_update_k(const float* __restrict__ act, const int* __restrict__ il,
                              const int* __restrict__ ir, const float* __restrict__ decay,
                              float* __restrict__ a, float* __restrict__ bsum,
                              float* __restrict__ syncv) {
  long idx = (long)blockIdx.x * blockDim.x + threadIdx.x;
  if (idx >= (long)BT_ * S_) return;
  int s  = (int)(idx % S_);
  long bt = idx / S_;
  float r  = expf(-decay[s]);
  float pa = act[bt * D_ + il[s]] * act[bt * D_ + ir[s]];
  float av = r * a[idx] + pa;
  float bv = r * bsum[idx] + 1.f;
  a[idx] = av; bsum[idx] = bv;
  syncv[idx] = av * rsqrtf(bv);
}

__global__ __launch_bounds__(256) void softmax_row256(float* __restrict__ s, float scale) {
  __shared__ float red[256];
  long row = blockIdx.x;
  float v = s[row * TT_ + threadIdx.x] * scale;
  float mx = block_red_max(v, red);
  float e = expf(v - mx);
  float sum = block_red_sum(e, red);
  s[row * TT_ + threadIdx.x] = e / sum;
}

__global__ void copy_act_to_xcat(const float* __restrict__ act, float* __restrict__ xcat) {
  long idx = (long)blockIdx.x * blockDim.x + threadIdx.x;
  if (idx >= (long)BT_ * D_) return;
  long row = idx / D_;
  int d = (int)(idx % D_);
  xcat[row * (2 * D_) + D_ + d] = act[idx];
}

// row LayerNorm (+optional SiLU), width 768, one block per row
__global__ __launch_bounds__(256) void ln_row_768(
    const float* __restrict__ in, float* __restrict__ out,
    const float* __restrict__ g, const float* __restrict__ b, int do_silu) {
  __shared__ float red[256];
  long row = blockIdx.x;
  const float* x = in + row * D_;
  float s = 0.f, s2 = 0.f;
  for (int i = threadIdx.x; i < D_; i += 256) { float t = x[i]; s += t; s2 += t * t; }
  s  = block_red_sum(s, red);
  s2 = block_red_sum(s2, red);
  float mean = s / (float)D_;
  float var  = fmaxf(s2 / (float)D_ - mean * mean, 0.f);
  float inv  = rsqrtf(var + 1e-5f);
  for (int i = threadIdx.x; i < D_; i += 256) {
    float y = (x[i] - mean) * inv * g[i] + b[i];
    if (do_silu) y = y * (1.f / (1.f + expf(-y)));
    out[row * D_ + i] = y;
  }
}

// row LayerNorm + SiLU, width 16, one thread per row
__global__ void ln_silu_w16(const float* __restrict__ in, float* __restrict__ out,
                            const float* __restrict__ g, const float* __restrict__ b) {
  int row = blockIdx.x * blockDim.x + threadIdx.x;
  if (row >= BT_) return;
  const float* x = in + (long)row * 16;
  float m = 0.f;
#pragma unroll
  for (int i = 0; i < 16; ++i) m += x[i];
  m *= (1.f / 16.f);
  float v = 0.f;
#pragma unroll
  for (int i = 0; i < 16; ++i) { float d = x[i] - m; v += d * d; }
  v *= (1.f / 16.f);
  float inv = rsqrtf(v + 1e-5f);
#pragma unroll
  for (int i = 0; i < 16; ++i) {
    float y = (x[i] - m) * inv * g[i] + b[i];
    out[(long)row * 16 + i] = y * (1.f / (1.f + expf(-y)));
  }
}

// state = LN(silu(LN(hupre,lnu)) + h0, lns)  — one block per row
__global__ __launch_bounds__(256) void synnet_tail(
    const float* __restrict__ hupre, const float* __restrict__ h0,
    float* __restrict__ state,
    const float* __restrict__ lnu_g, const float* __restrict__ lnu_b,
    const float* __restrict__ lns_g, const float* __restrict__ lns_b) {
  __shared__ float red[256];
  __shared__ float sb[D_];
  long row = blockIdx.x;
  const float* x = hupre + row * D_;
  float s = 0.f, s2 = 0.f;
  for (int i = threadIdx.x; i < D_; i += 256) { float t = x[i]; s += t; s2 += t * t; }
  s  = block_red_sum(s, red);
  s2 = block_red_sum(s2, red);
  float mean = s / (float)D_;
  float var  = fmaxf(s2 / (float)D_ - mean * mean, 0.f);
  float inv  = rsqrtf(var + 1e-5f);
  for (int i = threadIdx.x; i < D_; i += 256) {
    float y = (x[i] - mean) * inv * lnu_g[i] + lnu_b[i];
    y = y * (1.f / (1.f + expf(-y)));
    sb[i] = y + h0[row * D_ + i];
  }
  __syncthreads();
  s = 0.f; s2 = 0.f;
  for (int i = threadIdx.x; i < D_; i += 256) { float t = sb[i]; s += t; s2 += t * t; }
  s  = block_red_sum(s, red);
  s2 = block_red_sum(s2, red);
  float mean2 = s / (float)D_;
  float var2  = fmaxf(s2 / (float)D_ - mean2 * mean2, 0.f);
  float inv2  = rsqrtf(var2 + 1e-5f);
  for (int i = threadIdx.x; i < D_; i += 256)
    state[row * D_ + i] = (sb[i] - mean2) * inv2 * lns_g[i] + lns_b[i];
}

// trace shift + NLM1(LN25 + 25x8 matvec + GLU) + NLM2(LN4 + 4x2 matvec + GLU)
__global__ void nlm_step(float* __restrict__ trace, const float* __restrict__ state,
    const float* __restrict__ ln1g, const float* __restrict__ ln1b,
    const float* __restrict__ w1, const float* __restrict__ b1, const float* __restrict__ T1,
    const float* __restrict__ ln2g, const float* __restrict__ ln2b,
    const float* __restrict__ w2, const float* __restrict__ b2, const float* __restrict__ T2,
    float* __restrict__ act) {
  long idx = (long)blockIdx.x * blockDim.x + threadIdx.x;
  if (idx >= (long)BT_ * D_) return;
  int d = (int)(idx % D_);
  float* tp = trace + idx * MTR_;
  float tr[MTR_];
#pragma unroll
  for (int m = 0; m < MTR_ - 1; ++m) tr[m] = tp[m + 1];
  tr[MTR_ - 1] = state[idx];
#pragma unroll
  for (int m = 0; m < MTR_; ++m) tp[m] = tr[m];

  float mean = 0.f;
#pragma unroll
  for (int m = 0; m < MTR_; ++m) mean += tr[m];
  mean *= (1.f / MTR_);
  float var = 0.f;
#pragma unroll
  for (int m = 0; m < MTR_; ++m) { float dm = tr[m] - mean; var += dm * dm; }
  var *= (1.f / MTR_);
  float inv = rsqrtf(var + 1e-5f);
  float tn[MTR_];
#pragma unroll
  for (int m = 0; m < MTR_; ++m) tn[m] = (tr[m] - mean) * inv * ln1g[m] + ln1b[m];

  float invT1 = 1.f / T1[0];
  float y[8];
#pragma unroll
  for (int h = 0; h < 8; ++h) {
    float acc = 0.f;
#pragma unroll
    for (int m = 0; m < MTR_; ++m) acc += tn[m] * w1[(long)(m * 8 + h) * D_ + d];
    y[h] = (acc + b1[(long)d * 8 + h]) * invT1;
  }
  float hh[4];
#pragma unroll
  for (int j = 0; j < 4; ++j) hh[j] = y[j] * (1.f / (1.f + expf(-y[j + 4])));

  float m2 = 0.25f * (hh[0] + hh[1] + hh[2] + hh[3]);
  float v2 = 0.f;
#pragma unroll
  for (int j = 0; j < 4; ++j) { float dm = hh[j] - m2; v2 += dm * dm; }
  v2 *= 0.25f;
  float inv2 = rsqrtf(v2 + 1e-5f);
  float hn[4];
#pragma unroll
  for (int j = 0; j < 4; ++j) hn[j] = (hh[j] - m2) * inv2 * ln2g[j] + ln2b[j];

  float invT2 = 1.f / T2[0];
  float z0 = 0.f, z1 = 0.f;
#pragma unroll
  for (int h = 0; h < 4; ++h) {
    z0 += hn[h] * w2[(long)(h * 2 + 0) * D_ + d];
    z1 += hn[h] * w2[(long)(h * 2 + 1) * D_ + d];
  }
  z0 = (z0 + b2[(long)d * 2 + 0]) * invT2;
  z1 = (z1 + b2[(long)d * 2 + 1]) * invT2;
  act[idx] = z0 * (1.f / (1.f + expf(-z1)));
}

// LN over V + logits + log-softmax entropy; one block per (b,t) row
__global__ __launch_bounds__(256) void lm_finish(
    const float* __restrict__ pred, const float* __restrict__ gpt,
    const float* __restrict__ lnvg, const float* __restrict__ lnvb,
    const float* __restrict__ swp, const float* __restrict__ csp,
    float* __restrict__ out, int iter) {
  __shared__ float red[256];
  const long row = blockIdx.x;
  const float* p  = pred + row * (long)V_;
  const float* gl = gpt  + row * (long)V_;
  float* orow = out + row * (long)V_ * ITERS_ + iter;

  float s = 0.f, s2 = 0.f;
  for (int v = threadIdx.x; v < V_; v += 256) { float x = p[v]; s += x; s2 += x * x; }
  s  = block_red_sum(s, red);
  s2 = block_red_sum(s2, red);
  float mean = s / (float)V_;
  float var  = fmaxf(s2 / (float)V_ - mean * mean, 0.f);
  float inv  = rsqrtf(var + 1e-5f);
  float sw = swp[0], cs = csp[0];

  float mx = -3.4e38f;
  for (int v = threadIdx.x; v < V_; v += 256) {
    float lg = sw * gl[v] + cs * ((p[v] - mean) * inv * lnvg[v] + lnvb[v]);
    orow[(long)v * ITERS_] = lg;
    mx = fmaxf(mx, lg);
  }
  mx = block_red_max(mx, red);

  float S1 = 0.f, S2 = 0.f;
  for (int v = threadIdx.x; v < V_; v += 256) {
    float u = orow[(long)v * ITERS_] - mx;
    float e = expf(u);
    S1 += e; S2 += e * u;
  }
  S1 = block_red_sum(S1, red);
  S2 = block_red_sum(S2, red);
  if (threadIdx.x == 0) {
    float ent = logf(S1) - S2 / S1;                // -sum(p*logp)
    float ne  = ent / logf((float)V_);
    float* c = out + (long)BT_ * V_ * ITERS_ + row * 2 * ITERS_ + iter;
    c[0] = ne;
    c[ITERS_] = 1.f - ne;
  }
}

// ---------------------------------------------------------------------------
// Host orchestration.  Input indices assume the harness flattens
// setup_inputs() (and the nested params dict) in dict insertion order.
// ---------------------------------------------------------------------------
extern "C" void kernel_launch(void* const* d_in, const int* in_sizes, int n_in,
                              void* d_out, int out_size, void* d_ws, size_t ws_size,
                              hipStream_t stream) {
  (void)in_sizes; (void)n_in; (void)out_size; (void)ws_size;

  const float* kv       = (const float*)d_in[0];
  const float* gptlog   = (const float*)d_in[1];
  const float* syn_w_in = (const float*)d_in[2];
  const float* syn_b_in = (const float*)d_in[3];
  const float* ln0_g    = (const float*)d_in[4];
  const float* ln0_b    = (const float*)d_in[5];
  const float* w_down   = (const float*)d_in[6];
  const float* b_down   = (const float*)d_in[7];
  const float* lnd_g    = (const float*)d_in[8];
  const float* lnd_b    = (const float*)d_in[9];
  const float* w_up     = (const float*)d_in[10];
  const float* b_up     = (const float*)d_in[11];
  const float* lnu_g    = (const float*)d_in[12];
  const float* lnu_b    = (const float*)d_in[13];
  const float* lns_g    = (const float*)d_in[14];
  const float* lns_b    = (const float*)d_in[15];
  const float* n1_lng   = (const float*)d_in[16];
  const float* n1_lnb   = (const float*)d_in[17];
  const float* n1_w1    = (const float*)d_in[18];
  const float* n1_b1    = (const float*)d_in[19];
  const float* n1_T     = (const float*)d_in[20];
  const float* n2_lng   = (const float*)d_in[21];
  const float* n2_lnb   = (const float*)d_in[22];
  const float* n2_w1    = (const float*)d_in[23];
  const float* n2_b1    = (const float*)d_in[24];
  const float* n2_T     = (const float*)d_in[25];
  const float* q_w      = (const float*)d_in[26];
  const float* q_b      = (const float*)d_in[27];
  const float* k_w      = (const float*)d_in[28];
  const float* k_b      = (const float*)d_in[29];
  const float* v_w      = (const float*)d_in[30];
  const float* v_b      = (const float*)d_in[31];
  const float* out_w    = (const float*)d_in[32];
  const float* out_b    = (const float*)d_in[33];
  const float* lm_w     = (const float*)d_in[34];
  const float* lnv_g    = (const float*)d_in[35];
  const float* lnv_b    = (const float*)d_in[36];
  const float* dec_a    = (const float*)d_in[37];
  const float* dec_o    = (const float*)d_in[38];
  const float* start_a  = (const float*)d_in[39];
  const float* start_tr = (const float*)d_in[40];
  const float* skip_w   = (const float*)d_in[41];
  const float* ctm_s    = (const float*)d_in[42];
  const int* idx_al     = (const int*)d_in[43];
  const int* idx_ar     = (const int*)d_in[44];
  const int* idx_ol     = (const int*)d_in[45];
  const int* idx_or     = (const int*)d_in[46];

  // ---- workspace carve-up ----
  char* basep = (char*)d_ws;
  size_t off = 0;
  auto alloc = [&](size_t bytes) -> void* {
    void* p = basep + off;
    off += (bytes + 255) & ~(size_t)255;
    return p;
  };
  // all weight f16 buffers are TRANSPOSED: stored [N, K]
  _Float16* lmw16 = (_Float16*)alloc(2ull * D_ * V_);       // [V, D]
  _Float16* win16 = (_Float16*)alloc(2ull * 2 * D_ * D_);   // [D, 2D]
  _Float16* qw16  = (_Float16*)alloc(2ull * S_ * D_);       // [D, S]
  _Float16* kw16  = (_Float16*)alloc(2ull * D_ * D_);       // [D, D]
  _Float16* vw16  = (_Float16*)alloc(2ull * D_ * D_);       // [D, D]
  _Float16* ow16  = (_Float16*)alloc(2ull * S_ * D_);       // [D, S]
  _Float16* wd16  = (_Float16*)alloc(2ull * D_ * 16);       // [16, D]
  _Float16* wu16  = (_Float16*)alloc(2ull * 16 * D_);       // [D, 16]
  _Float16* kf16  = (_Float16*)alloc(2ull * BT_ * D_);      // [B*T, D]
  _Float16* vf16  = (_Float16*)alloc(2ull * BT_ * D_);      // [B*T, D]
  _Float16* vT16  = (_Float16*)alloc(2ull * BT_ * D_);      // [B, D, T]
  float* act    = (float*)alloc(4ull * BT_ * D_);
  float* trace  = (float*)alloc(4ull * BT_ * D_ * MTR_);
  float* aa     = (float*)alloc(4ull * BT_ * S_);
  float* ba     = (float*)alloc(4ull * BT_ * S_);
  float* ao     = (float*)alloc(4ull * BT_ * S_);
  float* bo     = (float*)alloc(4ull * BT_ * S_);
  float* syncb  = (float*)alloc(4ull * BT_ * S_);
  float* qbuf   = (float*)alloc(4ull * BT_ * D_);
  float* sbuf   = (float*)alloc(4ull * BT_ * TT_);
  float* xcat   = (float*)alloc(4ull * BT_ * 2 * D_);
  float* h0pre  = (float*)alloc(4ull * BT_ * D_);
  float* h0     = (float*)alloc(4ull * BT_ * D_);
  float* hdpre  = (float*)alloc(4ull * BT_ * 16);
  float* hd     = (float*)alloc(4ull * BT_ * 16);
  float* hupre  = (float*)alloc(4ull * BT_ * D_);
  float* state  = (float*)alloc(4ull * BT_ * D_);
  float* opb    = (float*)alloc(4ull * BT_ * D_);
  float* pred   = (float*)alloc(4ull * BT_ * V_);
  float* outp   = (float*)d_out;

  auto convT = [&](const float* src, _Float16* dst, int K, int N) {
    dim3 grid((unsigned)((N + 31) / 32), (unsigned)((K + 31) / 32));
    cvt_f16_T<<<grid, 256, 0, stream>>>(src, dst, K, N);
  };
  auto gemm = [&](const float* A, const _Float16* B, const float* bias,
                  float* C, _Float16* Ch, int M, int N, int K,
                  int lda, int ldb, int ldc,
                  long strA, long strB, long strC, int batch) {
    dim3 grid((unsigned)((N + BN - 1) / BN), (unsigned)((M + BM - 1) / BM),
              (unsigned)batch);
    wmma_gemm<<<grid, 256, 0, stream>>>(A, B, bias, C, Ch, M, N, K,
                                        lda, ldb, ldc, strA, strB, strC);
  };

  // ---- per-launch transposing weight conversions to f16 [N,K] ----
  convT(lm_w,     lmw16, D_, V_);        // [768,32000] -> [32000,768]
  convT(syn_w_in, win16, 2 * D_, D_);    // [1536,768]  -> [768,1536]
  convT(q_w,      qw16,  S_, D_);        // [384,768]   -> [768,384]
  convT(k_w,      kw16,  D_, D_);
  convT(v_w,      vw16,  D_, D_);
  convT(out_w,    ow16,  S_, D_);
  convT(w_down,   wd16,  D_, 16);        // [768,16]    -> [16,768]
  convT(w_up,     wu16,  16, D_);        // [16,768]    -> [768,16]

  // ---- carry init ----
  init_state_k<<<(BT_ * D_ + 255) / 256, 256, 0, stream>>>(start_a, start_tr, act, trace);
  init_sync_k<<<(BT_ * S_ + 255) / 256, 256, 0, stream>>>(start_a, idx_al, idx_ar, aa, ba);
  init_sync_k<<<(BT_ * S_ + 255) / 256, 256, 0, stream>>>(start_a, idx_ol, idx_or, ao, bo);

  // ---- k/v projections (f16 outputs; only consumed as WMMA B operands) ----
  gemm(kv, kw16, k_b, nullptr, kf16, BT_, D_, D_, D_, D_, D_, 0, 0, 0, 1);
  gemm(kv, vw16, v_b, nullptr, vf16, BT_, D_, D_, D_, D_, D_, 0, 0, 0, 1);
  transpose_v_f16<<<dim3(D_ / 32, TT_ / 32, 2), 256, 0, stream>>>(vf16, vT16);

  const float smscale = 1.0f / sqrtf((float)D_);

  for (int it = 0; it < ITERS_; ++it) {
    // action sync + attention
    sync_update_k<<<(BT_ * S_ + 255) / 256, 256, 0, stream>>>(
        act, idx_al, idx_ar, dec_a, aa, ba, syncb);
    gemm(syncb, qw16, q_b, qbuf, nullptr, BT_, D_, S_, S_, S_, D_, 0, 0, 0, 1);
    // scores = q @ k^T  (batched over B=2; kf16 is [T,D] == [N,K])
    gemm(qbuf, kf16, nullptr, sbuf, nullptr, TT_, TT_, D_, D_, D_, TT_,
         (long)TT_ * D_, (long)TT_ * D_, (long)TT_ * TT_, 2);
    softmax_row256<<<BT_, 256, 0, stream>>>(sbuf, smscale);
    // attn = P @ v  -> left half of xcat (ldc = 2D); vT16 is [D,T] == [N,K]
    gemm(sbuf, vT16, nullptr, xcat, nullptr, TT_, D_, TT_, TT_, TT_, 2 * D_,
         (long)TT_ * TT_, (long)D_ * TT_, (long)TT_ * 2 * D_, 2);
    copy_act_to_xcat<<<(BT_ * D_ + 255) / 256, 256, 0, stream>>>(act, xcat);

    // synapse UNet
    gemm(xcat, win16, syn_b_in, h0pre, nullptr, BT_, D_, 2 * D_,
         2 * D_, 2 * D_, D_, 0, 0, 0, 1);
    ln_row_768<<<BT_, 256, 0, stream>>>(h0pre, h0, ln0_g, ln0_b, 1);
    gemm(h0, wd16, b_down, hdpre, nullptr, BT_, 16, D_, D_, D_, 16, 0, 0, 0, 1);
    ln_silu_w16<<<(BT_ + 255) / 256, 256, 0, stream>>>(hdpre, hd, lnd_g, lnd_b);
    gemm(hd, wu16, b_up, hupre, nullptr, BT_, D_, 16, 16, 16, D_, 0, 0, 0, 1);
    synnet_tail<<<BT_, 256, 0, stream>>>(hupre, h0, state, lnu_g, lnu_b, lns_g, lns_b);

    // trace shift + per-neuron NLM stack -> new act
    nlm_step<<<(BT_ * D_ + 255) / 256, 256, 0, stream>>>(
        trace, state, n1_lng, n1_lnb, n1_w1, n1_b1, n1_T,
        n2_lng, n2_lnb, n2_w1, n2_b1, n2_T, act);

    // output sync + LM head
    sync_update_k<<<(BT_ * S_ + 255) / 256, 256, 0, stream>>>(
        act, idx_ol, idx_or, dec_o, ao, bo, syncb);
    gemm(syncb, ow16, out_b, opb, nullptr, BT_, D_, S_, S_, S_, D_, 0, 0, 0, 1);
    gemm(opb, lmw16, nullptr, pred, nullptr, BT_, V_, D_, D_, D_, V_, 0, 0, 0, 1);
    lm_finish<<<BT_, 256, 0, stream>>>(pred, gptlog, lnv_g, lnv_b, skip_w, ctm_s, outp, it);
  }
}
